// MoeLayer_35596688949260
// MI455X (gfx1250) — matrix operand order repaired and checked by reading, not compiled, not run on hardware
//
#include <hip/hip_runtime.h>
#include <hip/hip_bf16.h>
#include <stdint.h>

// Problem constants (from reference): B=4, S=2048, D=2048, O=2048, E=8, top-k=2
#define D_DIM 2048
#define O_DIM 2048
#define E_NUM 8
#define T_NUM 8192          // B*S tokens
#define CAP   8192          // per-expert bucket capacity (worst case = T)

typedef __bf16 bf16_t;
typedef __attribute__((ext_vector_type(16))) __bf16 v16bf;
typedef __attribute__((ext_vector_type(8)))  float  v8f;

// Workspace layout (bytes)
#define OFF_WB   ((size_t)0)                                  // bf16 expert weights: 64 MiB
#define OFF_XB   ((size_t)E_NUM * O_DIM * D_DIM * 2)          // bf16 activations:   32 MiB
#define OFF_CNT  (OFF_XB + (size_t)T_NUM * D_DIM * 2)         // 8 ints
#define OFF_TOK  (OFF_CNT + 256)                              // E*CAP ints   (bucketed token ids)
#define OFF_BW   (OFF_TOK + (size_t)E_NUM * CAP * 4)          // E*CAP floats (gate weights)

__device__ __forceinline__ unsigned f2bf_bits(float f) {
    unsigned u = __float_as_uint(f);
    return (u + 0x7FFFu + ((u >> 16) & 1u)) >> 16;   // round-to-nearest-even
}

// ---------------------------------------------------------------------------
// Kernel 0: zero output accumulator + expert counters
// ---------------------------------------------------------------------------
__global__ void k_zero(float* __restrict__ out, size_t n4, int* __restrict__ cnt) {
    size_t i = (size_t)blockIdx.x * blockDim.x + threadIdx.x;
    size_t stride = (size_t)gridDim.x * blockDim.x;
    float4 z = make_float4(0.f, 0.f, 0.f, 0.f);
    for (; i < n4; i += stride) ((float4*)out)[i] = z;
    if (blockIdx.x == 0 && threadIdx.x < E_NUM) cnt[threadIdx.x] = 0;
}

// ---------------------------------------------------------------------------
// Kernel 1: fp32 -> bf16 conversion (used for expert weights AND activations;
// both bf16 copies together (~100MB) fit in the 192MB global L2)
// ---------------------------------------------------------------------------
__global__ void k_cvt(const float* __restrict__ w, unsigned* __restrict__ o, int n2) {
    int i = blockIdx.x * blockDim.x + threadIdx.x;
    int stride = gridDim.x * blockDim.x;
    for (; i < n2; i += stride) {
        float2 f = ((const float2*)w)[i];
        o[i] = f2bf_bits(f.x) | (f2bf_bits(f.y) << 16);
    }
}

// ---------------------------------------------------------------------------
// Kernel 2: gating + routing. One wave32 per token: 8 expert logits via
// lane-strided dot products (full fp32 precision), shfl reduce, top-2
// softmax, scatter into fixed-capacity per-expert buckets.
// ---------------------------------------------------------------------------
__global__ __launch_bounds__(256)
void k_gate(const float* __restrict__ x, const float* __restrict__ gw,
            int* __restrict__ cnt, int* __restrict__ btok, float* __restrict__ bw) {
    int wid  = threadIdx.x >> 5;
    int lane = threadIdx.x & 31;
    int t = blockIdx.x * 8 + wid;
    const float* xr = x + (size_t)t * D_DIM;

    float acc[E_NUM];
#pragma unroll
    for (int e = 0; e < E_NUM; ++e) acc[e] = 0.f;

    for (int d = lane; d < D_DIM; d += 32) {
        float xv = xr[d];
#pragma unroll
        for (int e = 0; e < E_NUM; ++e) acc[e] += xv * gw[e * D_DIM + d];
    }
#pragma unroll
    for (int e = 0; e < E_NUM; ++e) {
#pragma unroll
        for (int m = 16; m >= 1; m >>= 1) acc[e] += __shfl_xor(acc[e], m, 32);
    }
    if (lane == 0) {
        int e0 = 0; float b0 = acc[0];
#pragma unroll
        for (int e = 1; e < E_NUM; ++e) if (acc[e] > b0) { b0 = acc[e]; e0 = e; }
        int e1 = (e0 == 0) ? 1 : 0; float b1 = acc[e1];
#pragma unroll
        for (int e = 0; e < E_NUM; ++e)
            if (e != e0 && acc[e] > b1) { b1 = acc[e]; e1 = e; }
        float q  = __expf(b1 - b0);          // softmax over [b0, b1], b0 = max
        float w0 = 1.f / (1.f + q);
        float w1 = 1.f - w0;
        int p0 = atomicAdd(&cnt[e0], 1);
        btok[e0 * CAP + p0] = t; bw[e0 * CAP + p0] = w0;
        int p1 = atomicAdd(&cnt[e1], 1);
        btok[e1 * CAP + p1] = t; bw[e1 * CAP + p1] = w1;
    }
}

// ---------------------------------------------------------------------------
// Kernel 3: grouped expert GEMM, LDS-free / barrier-free.
//   Block = 256 threads = 8 wave32s; block tile M=64 x N=256.
//   Wave (mg = wid&1, ng = wid>>1) computes a 32x64 C block:
//     2 A fragments x 4 B fragments -> 8 WMMAs per 32-K step,
//     fed by 12 global_load_b128 (all L2-resident bf16 data).
//   A layout (16-bit A 16x32): lanes 0-15 K{0-7,16-23}, lanes 16-31 K{8-15,24-31}
//   B layout (16-bit B 32x16): lanes 0-15 K{0-7,16-23}, lanes 16-31 K{8-15,24-31}
// ---------------------------------------------------------------------------
__global__ __launch_bounds__(256)
void k_gemm(const bf16_t* __restrict__ xb, const bf16_t* __restrict__ wb,
            const float* __restrict__ eb, const int* __restrict__ cnt,
            const int* __restrict__ btok, const float* __restrict__ bw,
            float* __restrict__ out) {
    int e     = blockIdx.y >> 7;          // CAP/64 = 128 row tiles per expert
    int tloc  = blockIdx.y & 127;
    int rows  = cnt[e];
    int rbase = tloc * 64;
    if (rbase >= rows) return;            // uniform early-exit

    int tid  = threadIdx.x;
    int wid  = tid >> 5;
    int lane = tid & 31;
    int half = lane >> 4;
    int n    = lane & 15;
    int mg   = wid & 1;                   // M group: rows 0-31 / 32-63
    int ng   = wid >> 1;                  // N group: 64 cols each

    // ---- A per-lane source rows (tokens gathered from bucket) ----
    int rr0  = rbase + mg * 32 + (lane & 15);
    int rr1  = rr0 + 16;
    int tok0 = (rr0 < rows) ? btok[e * CAP + rr0] : 0;
    int tok1 = (rr1 < rows) ? btok[e * CAP + rr1] : 0;
    const bf16_t* a0 = xb + (size_t)tok0 * D_DIM + half * 8;
    const bf16_t* a1 = xb + (size_t)tok1 * D_DIM + half * 8;

    // ---- B per-lane sources: 4 column tiles, K contiguous in memory ----
    int obase = blockIdx.x * 256 + ng * 64 + n;
    const bf16_t* wr[4];
#pragma unroll
    for (int u = 0; u < 4; ++u)
        wr[u] = wb + ((size_t)e * O_DIM + obase + u * 16) * D_DIM + half * 8;

    v8f c[2][4];
#pragma unroll
    for (int t = 0; t < 2; ++t)
#pragma unroll
        for (int u = 0; u < 4; ++u)
            c[t][u] = (v8f){0.f, 0.f, 0.f, 0.f, 0.f, 0.f, 0.f, 0.f};

#pragma unroll 2
    for (int kb = 0; kb < D_DIM; kb += 32) {
        union { uint4 q[2]; v16bf v; } A0, A1, B[4];
        A0.q[0] = *(const uint4*)(a0 + kb);
        A0.q[1] = *(const uint4*)(a0 + kb + 16);
        A1.q[0] = *(const uint4*)(a1 + kb);
        A1.q[1] = *(const uint4*)(a1 + kb + 16);
#pragma unroll
        for (int u = 0; u < 4; ++u) {
            B[u].q[0] = *(const uint4*)(wr[u] + kb);
            B[u].q[1] = *(const uint4*)(wr[u] + kb + 16);
        }
        // Unconditional near-cache prefetch of the next K-slab (speculative
        // prefetch is drop-safe; stays within the ws allocation regardless).
        __builtin_prefetch(wr[0] + kb + 64, 0, 3);
        __builtin_prefetch(wr[2] + kb + 64, 0, 3);
#pragma unroll
        for (int u = 0; u < 4; ++u) {
            c[0][u] = __builtin_amdgcn_wmma_f32_16x16x32_bf16(
                          false, A0.v, false, B[u].v, (short)0, c[0][u], false, false);
            c[1][u] = __builtin_amdgcn_wmma_f32_16x16x32_bf16(
                          false, A1.v, false, B[u].v, (short)0, c[1][u], false, false);
        }
    }

    // ---- Epilogue: bias + gate-weight scale, atomic f32 accumulate ----
    float bias[4];
#pragma unroll
    for (int u = 0; u < 4; ++u) bias[u] = eb[e * O_DIM + obase + u * 16];

#pragma unroll
    for (int t = 0; t < 2; ++t) {
#pragma unroll
        for (int i = 0; i < 8; ++i) {
            int m = mg * 32 + t * 16 + half * 8 + i;   // row within block tile
            int r = rbase + m;
            if (r < rows) {
                int   tk  = btok[e * CAP + r];
                float wgt = bw[e * CAP + r];
                float* orow = out + (size_t)tk * O_DIM;
#pragma unroll
                for (int u = 0; u < 4; ++u)
                    unsafeAtomicAdd(&orow[obase + u * 16],
                                    wgt * (c[t][u][i] + bias[u]));
            }
        }
    }
}

// ---------------------------------------------------------------------------
extern "C" void kernel_launch(void* const* d_in, const int* in_sizes, int n_in,
                              void* d_out, int out_size, void* d_ws, size_t ws_size,
                              hipStream_t stream) {
    const float* x  = (const float*)d_in[0];   // (B,S,D)
    const float* gw = (const float*)d_in[1];   // (E,D)
    const float* ew = (const float*)d_in[2];   // (E,O,D)
    const float* eb = (const float*)d_in[3];   // (E,O)
    float* out = (float*)d_out;

    char* ws = (char*)d_ws;
    bf16_t* wb   = (bf16_t*)(ws + OFF_WB);
    bf16_t* xb   = (bf16_t*)(ws + OFF_XB);
    int*    cnt  = (int*)   (ws + OFF_CNT);
    int*    btok = (int*)   (ws + OFF_TOK);
    float*  bw   = (float*) (ws + OFF_BW);

    // 1) zero output + counters
    k_zero<<<1024, 256, 0, stream>>>(out, (size_t)T_NUM * O_DIM / 4, cnt);
    // 2) weights + activations fp32 -> bf16 (single pass each)
    k_cvt<<<4096, 256, 0, stream>>>(ew, (unsigned*)wb, E_NUM * O_DIM * D_DIM / 2);
    k_cvt<<<2048, 256, 0, stream>>>(x,  (unsigned*)xb, T_NUM * D_DIM / 2);
    // 3) gating + routing scatter (fp32 precision)
    k_gate<<<T_NUM / 8, 256, 0, stream>>>(x, gw, cnt, btok, bw);
    // 4) grouped expert GEMM (WMMA bf16), atomic f32 accumulate into out
    dim3 g(O_DIM / 256, E_NUM * (CAP / 64), 1);
    k_gemm<<<g, 256, 0, stream>>>(xb, wb, eb, cnt, btok, bw, out);
}
